// SwinTransformerBlock_64381559767407
// MI455X (gfx1250) — matrix operand order repaired and checked by reading, not compiled, not run on hardware
//
#include <hip/hip_runtime.h>
#include <hip/hip_bf16.h>

typedef __bf16 bf16;
typedef __attribute__((ext_vector_type(16))) __bf16 v16bf;
typedef __attribute__((ext_vector_type(8)))  __bf16 v8bf;
typedef __attribute__((ext_vector_type(8)))  float  v8f;

#define NUMH  6
#define WSZ   7
#define SHIFT 3
#define CH    192
#define TOK   49
#define TPAD  64
#define HW    112

// native bf16 convert (clang lowers to v_cvt_*bf16_f32 on gfx1250)
__device__ __forceinline__ bf16 to_bf16(float f) { return (bf16)f; }

// --- prep: f32 weights -> bf16, transposed to [head][n][k] ---------------
// Wq/Wk/Wv: [C][H][C] (k,h,n) -> WT[h][n][k]
// Wo:       [H][C][C] (h,d,c) -> WoT[h][c][d]
__global__ void swin_prep_weights(const float* __restrict__ Wq,
                                  const float* __restrict__ Wk,
                                  const float* __restrict__ Wv,
                                  const float* __restrict__ Wo,
                                  bf16* __restrict__ ws) {
  const int PER = NUMH * CH * CH;
  int idx = blockIdx.x * blockDim.x + threadIdx.x;
  if (idx >= PER) return;
  int h = idx / (CH * CH);
  int r = idx - h * (CH * CH);
  int n = r / CH;
  int k = r - n * CH;
  int src = k * (NUMH * CH) + h * CH + n;
  ws[idx]           = to_bf16(Wq[src]);
  ws[PER + idx]     = to_bf16(Wk[src]);
  ws[2 * PER + idx] = to_bf16(Wv[src]);
  ws[3 * PER + idx] = to_bf16(Wo[h * (CH * CH) + k * CH + n]);
}

// --- fragment loaders (wave32 bf16 WMMA layouts, ISA 7.12.2) -------------
// A 16x32: lane m = lane&15; elems 0..7 -> K = (lane>>4)*8 + i,
//          elems 8..15 -> K = 16 + (lane>>4)*8 + (i-8)
__device__ __forceinline__ v16bf load_a_frag(const bf16* base, int ld,
                                             int m0, int k0, int lane) {
  int m  = m0 + (lane & 15);
  int kh = (lane >> 4) << 3;
  const bf16* p = base + m * ld + k0 + kh;
  v8bf lo = *(const v8bf*)(p);
  v8bf hi = *(const v8bf*)(p + 16);
  v16bf f;
#pragma unroll
  for (int i = 0; i < 8; ++i) { f[i] = lo[i]; f[i + 8] = hi[i]; }
  return f;
}

// B 32x16 from row-major [N][K] storage (i.e. data[n*ld+k] == B[k][n]):
// lane n = lane&15; elems 0..7 -> K = (lane>>4)*8 + i, elems 8..15 -> K = 16 + (lane>>4)*8 + (i-8)
__device__ __forceinline__ v16bf load_bT_frag(const bf16* base, int ld,
                                              int n0, int k0, int lane) {
  int n  = n0 + (lane & 15);
  int kb = (lane >> 4) << 3;
  const bf16* p = base + n * ld + k0 + kb;
  v8bf lo = *(const v8bf*)(p);
  v8bf hi = *(const v8bf*)(p + 16);
  v16bf f;
#pragma unroll
  for (int i = 0; i < 8; ++i) { f[i] = lo[i]; f[i + 8] = hi[i]; }
  return f;
}

__device__ __forceinline__ v8f wmma_bf16(v16bf a, v16bf b, v8f c) {
  return __builtin_amdgcn_wmma_f32_16x16x32_bf16(false, a, false, b,
                                                 (short)0, c, false, false);
}

// --- main kernel: one workgroup (8 wave32) per 7x7 window ----------------
__global__ __launch_bounds__(256, 1)
void swin_attn_kernel(const float* __restrict__ x,
                      const float* __restrict__ bq,
                      const float* __restrict__ bk,
                      const float* __restrict__ bv,
                      const float* __restrict__ bo,
                      const bf16* __restrict__ wT,   // [WqT|WkT|WvT|WoT]
                      float* __restrict__ out) {
  extern __shared__ char smem[];
  bf16*  Xs  = (bf16*)(smem);            // [64][192] bf16
  bf16*  Qs  = (bf16*)(smem + 24576);    // [64][192]
  bf16*  Ks  = (bf16*)(smem + 49152);    // [64][192]
  bf16*  VTs = (bf16*)(smem + 73728);    // [192][64] (transposed V)
  bf16*  Ps  = (bf16*)(smem + 98304);    // [64][64] probs
  bf16*  Ts  = (bf16*)(smem + 106496);   // [64][192] P@V
  float* Ss  = (float*)(smem + 106496);  // [64][64] scores, aliases Ts

  const int w    = blockIdx.x;     // 0..4095
  const int b    = w >> 8;
  const int wrem = w & 255;
  const int wi   = wrem >> 4;
  const int wj   = wrem & 15;
  const int tid  = threadIdx.x;
  const int lane = tid & 31;
  const int wave = tid >> 5;
  const int PER  = NUMH * CH * CH;
  const float scale = 0.07216878364870323f; // 1/sqrt(192)

  // ---- Phase A: load shifted-window tile -> bf16 LDS, pad rows 49..63 ---
  for (int idx = tid; idx < TPAD * (CH / 4); idx += 256) {
    int r  = idx / (CH / 4);
    int c4 = (idx - r * (CH / 4)) * 4;
    float4 v = make_float4(0.f, 0.f, 0.f, 0.f);
    if (r < TOK) {
      int i = r / WSZ, j = r - i * WSZ;
      int gh = wi * WSZ + i + SHIFT; if (gh >= HW) gh -= HW;
      int gw = wj * WSZ + j + SHIFT; if (gw >= HW) gw -= HW;
      v = *(const float4*)(x + (((long)b * HW + gh) * HW + gw) * CH + c4);
    }
    // pack 4 bf16 (8B aligned store)
    bf16 pk[4] = { to_bf16(v.x), to_bf16(v.y), to_bf16(v.z), to_bf16(v.w) };
    *(unsigned long long*)(Xs + r * CH + c4) = *(unsigned long long*)pk;
  }
  __syncthreads();

  // persistent output accumulators: each wave owns 6 of the 4x12 tiles
  v8f acc[6];
#pragma unroll
  for (int u = 0; u < 6; ++u)
#pragma unroll
    for (int j = 0; j < 8; ++j) acc[u][j] = 0.f;

  for (int h = 0; h < NUMH; ++h) {
    // ---- B1: Q/K/V = X @ W_h (+bias); V stored transposed --------------
    for (int t = wave; t < 144; t += 8) {
      int mat = t / 48;
      int rem = t - mat * 48;
      int mt  = rem / 12, nt = rem - mt * 12;
      const bf16* WT = wT + mat * PER + h * (CH * CH);
      v8f d;
#pragma unroll
      for (int j = 0; j < 8; ++j) d[j] = 0.f;
#pragma unroll
      for (int kk = 0; kk < CH; kk += 32)
        d = wmma_bf16(load_a_frag(Xs, CH, mt * 16, kk, lane),
                      load_bT_frag(WT, CH, nt * 16, kk, lane), d);
      const float* bias = (mat == 0) ? bq : ((mat == 1) ? bk : bv);
      int n  = nt * 16 + (lane & 15);
      int mb = mt * 16 + ((lane >> 4) << 3);
      float bval = bias[h * CH + n];
      if (mat == 2) {
        // transposed V: 8 contiguous bf16 along tokens -> one 16B store
        bf16 pk[8];
#pragma unroll
        for (int j = 0; j < 8; ++j) pk[j] = to_bf16(d[j] + bval);
        *(v8bf*)(VTs + n * TPAD + mb) = *(v8bf*)pk;
      } else {
        bf16* dstm = (mat == 0) ? Qs : Ks;
#pragma unroll
        for (int j = 0; j < 8; ++j) dstm[(mb + j) * CH + n] = to_bf16(d[j] + bval);
      }
    }
    __syncthreads();

    // ---- B2: S = Q @ K^T * scale  (4x4 tiles) --------------------------
    for (int t = wave; t < 16; t += 8) {
      int mt = t >> 2, nt = t & 3;
      v8f d;
#pragma unroll
      for (int j = 0; j < 8; ++j) d[j] = 0.f;
#pragma unroll
      for (int kk = 0; kk < CH; kk += 32)
        d = wmma_bf16(load_a_frag(Qs, CH, mt * 16, kk, lane),
                      load_bT_frag(Ks, CH, nt * 16, kk, lane), d);
      int n  = nt * 16 + (lane & 15);
      int mb = mt * 16 + ((lane >> 4) << 3);
#pragma unroll
      for (int j = 0; j < 8; ++j) Ss[(mb + j) * TPAD + n] = d[j] * scale;
    }
    __syncthreads();

    // ---- B3: masked softmax over 49 valid keys, rows 49..63 zeroed -----
    if (tid < TPAD) {
      int trow = tid;
      if (trow < TOK) {
        float mx = -1e30f;
        for (int s = 0; s < TOK; ++s) mx = fmaxf(mx, Ss[trow * TPAD + s]);
        float sum = 0.f;
        for (int s = 0; s < TOK; ++s) sum += __expf(Ss[trow * TPAD + s] - mx);
        float inv = 1.f / sum;
        for (int s = 0; s < TOK; ++s)
          Ps[trow * TPAD + s] = to_bf16(__expf(Ss[trow * TPAD + s] - mx) * inv);
        for (int s = TOK; s < TPAD; ++s) Ps[trow * TPAD + s] = to_bf16(0.f);
      } else {
        for (int s = 0; s < TPAD; ++s) Ps[trow * TPAD + s] = to_bf16(0.f);
      }
    }
    __syncthreads();

    // ---- B4: T = P @ V  (4x12 tiles, K=64) -----------------------------
    for (int t = wave; t < 48; t += 8) {
      int mt = t / 12, nt = t - mt * 12;
      v8f d;
#pragma unroll
      for (int j = 0; j < 8; ++j) d[j] = 0.f;
#pragma unroll
      for (int kk = 0; kk < TPAD; kk += 32)
        d = wmma_bf16(load_a_frag(Ps, TPAD, mt * 16, kk, lane),
                      load_bT_frag(VTs, TPAD, nt * 16, kk, lane), d);
      int n  = nt * 16 + (lane & 15);
      int mb = mt * 16 + ((lane >> 4) << 3);
#pragma unroll
      for (int j = 0; j < 8; ++j) Ts[(mb + j) * CH + n] = to_bf16(d[j]);
    }
    __syncthreads();

    // ---- B5: acc += T @ Wo_h^T (persistent register accumulators) ------
    const bf16* WoTh = wT + 3 * PER + h * (CH * CH);
#pragma unroll
    for (int u = 0; u < 6; ++u) {
      int tt = wave * 6 + u;
      int mt = tt / 12, nt = tt - mt * 12;
#pragma unroll
      for (int kk = 0; kk < CH; kk += 32)
        acc[u] = wmma_bf16(load_a_frag(Ts, CH, mt * 16, kk, lane),
                           load_bT_frag(WoTh, CH, nt * 16, kk, lane), acc[u]);
    }
    __syncthreads();
  }

  // ---- store: +bias, plain-reshape window merge, roll(+3,+3) -----------
  const int wb_flat_base = wrem * (TOK * CH);
#pragma unroll
  for (int u = 0; u < 6; ++u) {
    int tt = wave * 6 + u;
    int mt = tt / 12, nt = tt - mt * 12;
    int n  = nt * 16 + (lane & 15);
    int mb = mt * 16 + ((lane >> 4) << 3);
    float bout = bo[n];
#pragma unroll
    for (int j = 0; j < 8; ++j) {
      int m = mb + j;
      if (m < TOK) {
        int flat = wb_flat_base + m * CH + n;     // pre-roll within-batch flat
        int hh = flat / (HW * CH);
        int ww = (flat - hh * (HW * CH)) / CH;
        hh += SHIFT; if (hh >= HW) hh -= HW;
        ww += SHIFT; if (ww >= HW) ww -= HW;
        out[(((long)b * HW + hh) * HW + ww) * CH + n] = acc[u][j] + bout;
      }
    }
  }
}

extern "C" void kernel_launch(void* const* d_in, const int* in_sizes, int n_in,
                              void* d_out, int out_size, void* d_ws, size_t ws_size,
                              hipStream_t stream) {
  const float* x  = (const float*)d_in[0];
  const float* Wq = (const float*)d_in[1];
  const float* bq = (const float*)d_in[2];
  const float* Wk = (const float*)d_in[3];
  const float* bk = (const float*)d_in[4];
  const float* Wv = (const float*)d_in[5];
  const float* bv = (const float*)d_in[6];
  const float* Wo = (const float*)d_in[7];
  const float* bo = (const float*)d_in[8];
  float* out = (float*)d_out;
  bf16* wT = (bf16*)d_ws;   // needs 4*6*192*192*2 = 1,769,472 bytes

  const int PER = NUMH * CH * CH;
  hipLaunchKernelGGL(swin_prep_weights, dim3((PER + 255) / 256), dim3(256), 0,
                     stream, Wq, Wk, Wv, Wo, wT);
  hipLaunchKernelGGL(swin_attn_kernel, dim3(4096), dim3(256), 131072,
                     stream, x, bq, bk, bv, bo, wT, out);
}
